// MinVQVAE_Cos_MultiQuery_12902081757259
// MI455X (gfx1250) — compile-verified, matrix operand order checked by reading, not run on hardware
//
#include <hip/hip_runtime.h>
#include <hip/hip_bf16.h>
#include <math.h>

typedef _Float16 h16;
typedef __attribute__((ext_vector_type(16))) _Float16 v16h;
typedef __attribute__((ext_vector_type(8)))  _Float16 v8h;
typedef __attribute__((ext_vector_type(4)))  _Float16 v4h;
typedef __attribute__((ext_vector_type(8)))  float    v8f;
typedef __attribute__((ext_vector_type(4)))  float    v4f;

#define SQRT1_2F 0.70710678118654752440f

// ---------------------------------------------------------------------------
// WMMA helpers (CDNA5 16x16x32 f16 -> f32)
// ---------------------------------------------------------------------------
__device__ __forceinline__ v8f wmma16(v16h a, v16h b, v8f c) {
  return __builtin_amdgcn_wmma_f32_16x16x32_f16(
      /*neg_a=*/false, a, /*neg_b=*/false, b,
      /*c_mod=*/(short)0, c, /*reuse_a=*/false, /*reuse_b=*/false);
}

// Load a 16x32 fragment (row-major [rows, ld]) for tile base `rc` at K offset
// `k0`.  Per ISA layout: lane holds row rc+(lane&15); lanes 0-15 cover
// K = k0+{0..7, 16..23}, lanes 16-31 cover K = k0+{8..15, 24..31}.  Both
// halves are contiguous 16-byte chunks -> two global_load_b128.  The same
// layout serves the B matrix because weights are stored [N, K].
__device__ __forceinline__ v16h load_frag(const h16* __restrict__ p, int ld,
                                          int rc, int k0, int lane) {
  const h16* base = p + (size_t)(rc + (lane & 15)) * (size_t)ld
                      + (size_t)(k0 + ((lane & 16) ? 8 : 0));
  v8h lo = *(const v8h*)(base);
  v8h hi = *(const v8h*)(base + 16);
  v16h r;
#pragma unroll
  for (int i = 0; i < 8; ++i) { r[i] = lo[i]; r[i + 8] = hi[i]; }
  return r;
}

// ---------------------------------------------------------------------------
// Fused GEMM: C[M,N] = act(A[M,K] * W[N,K]^T + bias[N])
// 256 threads = 8 wave32 (2x4); block tile 128x128; wave tile 64x32
// (4x2 WMMA tiles -> 12 b128 loads per 8 WMMAs, A reused 2x, B reused 4x).
// ACT: 0 = none, 1 = exact (erf) GELU, 2 = sigmoid.
// ---------------------------------------------------------------------------
template <int ACT, bool SF32, bool SF16>
__global__ __launch_bounds__(256) void gemm_wmma_kernel(
    const h16* __restrict__ A, const h16* __restrict__ W,
    const float* __restrict__ bias,
    float* __restrict__ outF, h16* __restrict__ outH,
    int M, int N, int K) {
  const int lane = threadIdx.x & 31;
  const int wave = threadIdx.x >> 5;
  const int m0 = blockIdx.y * 128 + (wave >> 2) * 64;  // 2 waves along M, 64 rows each
  const int n0 = blockIdx.x * 128 + (wave & 3) * 32;   // 4 waves along N, 32 cols each

  v8f acc[4][2] = {};
  for (int k0 = 0; k0 < K; k0 += 32) {
    v16h a0 = load_frag(A, K, m0,      k0, lane);
    v16h a1 = load_frag(A, K, m0 + 16, k0, lane);
    v16h a2 = load_frag(A, K, m0 + 32, k0, lane);
    v16h a3 = load_frag(A, K, m0 + 48, k0, lane);
    v16h b0 = load_frag(W, K, n0,      k0, lane);
    v16h b1 = load_frag(W, K, n0 + 16, k0, lane);
    acc[0][0] = wmma16(a0, b0, acc[0][0]);
    acc[0][1] = wmma16(a0, b1, acc[0][1]);
    acc[1][0] = wmma16(a1, b0, acc[1][0]);
    acc[1][1] = wmma16(a1, b1, acc[1][1]);
    acc[2][0] = wmma16(a2, b0, acc[2][0]);
    acc[2][1] = wmma16(a2, b1, acc[2][1]);
    acc[3][0] = wmma16(a3, b0, acc[3][0]);
    acc[3][1] = wmma16(a3, b1, acc[3][1]);
  }

  // C/D layout: VGPR v holds row (v + 8*(lane>=16)), column lane&15.
  const int col_in = lane & 15;
  const int rhalf  = (lane & 16) ? 8 : 0;
#pragma unroll
  for (int j = 0; j < 2; ++j) {
    const int col = n0 + j * 16 + col_in;
    const float bv = bias[col];
#pragma unroll
    for (int i = 0; i < 4; ++i) {
#pragma unroll
      for (int v = 0; v < 8; ++v) {
        const int row = m0 + i * 16 + rhalf + v;
        float x = acc[i][j][v] + bv;
        if (ACT == 1) x = 0.5f * x * (1.0f + erff(x * SQRT1_2F));
        else if (ACT == 2) x = 1.0f / (1.0f + __expf(-x));
        const size_t off = (size_t)row * (size_t)N + (size_t)col;
        if (SF32) outF[off] = x;
        if (SF16) outH[off] = (h16)x;
      }
    }
  }
}

// ---------------------------------------------------------------------------
// Codebook scoring + argmax:  factor[r, c] = <z_e[r, :256], pool[c, :256]>,
// z_index[r] = argmax_c.  16 rows per wave, 128 rows per block.
// A fragments (rows) are preloaded once; 64 code tiles x 8 WMMAs each.
// ---------------------------------------------------------------------------
__global__ __launch_bounds__(256) void score_argmax_kernel(
    const h16* __restrict__ ZE,    // [R, 256] f16
    const h16* __restrict__ POOL,  // [1024, 256] f16
    int* __restrict__ zidx, int R) {
  const int lane = threadIdx.x & 31;
  const int wave = threadIdx.x >> 5;
  const int r0 = blockIdx.x * 128 + wave * 16;
  if (r0 >= R) return;

  v16h afrag[8];
#pragma unroll
  for (int kk = 0; kk < 8; ++kk) afrag[kk] = load_frag(ZE, 256, r0, kk * 32, lane);

  float bestV[8];
  int   bestI[8];
#pragma unroll
  for (int v = 0; v < 8; ++v) { bestV[v] = -3.402823466e38f; bestI[v] = 0; }

  const int col_in = lane & 15;
  for (int c0 = 0; c0 < 1024; c0 += 16) {
    v8f acc = {};
#pragma unroll
    for (int kk = 0; kk < 8; ++kk) {
      v16h b = load_frag(POOL, 256, c0, kk * 32, lane);
      acc = wmma16(afrag[kk], b, acc);
    }
    const int col = c0 + col_in;
#pragma unroll
    for (int v = 0; v < 8; ++v) {
      float val = acc[v];
      if (val > bestV[v]) { bestV[v] = val; bestI[v] = col; }  // strict > keeps first
    }
  }

  // Cross-lane argmax within each 16-lane half (columns); first-index tiebreak.
#pragma unroll
  for (int v = 0; v < 8; ++v) {
    float bv = bestV[v];
    int   bi = bestI[v];
#pragma unroll
    for (int m = 1; m < 16; m <<= 1) {
      float ov = __shfl_xor(bv, m, 32);
      int   oi = __shfl_xor(bi, m, 32);
      if (ov > bv || (ov == bv && oi < bi)) { bv = ov; bi = oi; }
    }
    if (col_in == 0) {
      const int row = r0 + v + ((lane & 16) ? 8 : 0);
      zidx[row] = bi;
    }
  }
}

// ---------------------------------------------------------------------------
// Elementwise / reduction utility kernels
// ---------------------------------------------------------------------------
__device__ __forceinline__ void wave_atomic_add(float* dst, float s) {
#pragma unroll
  for (int m = 16; m >= 1; m >>= 1) s += __shfl_xor(s, m, 32);
  if ((threadIdx.x & 31) == 0) atomicAdd(dst, s);
}

// Vectorized f32 -> f16 (4 elements/thread/iter; n is a multiple of 4).
__global__ __launch_bounds__(256) void cvt_f32_to_f16(
    const float* __restrict__ in, h16* __restrict__ out, size_t n4) {
  size_t i = (size_t)blockIdx.x * blockDim.x + threadIdx.x;
  const size_t stride = (size_t)gridDim.x * blockDim.x;
  const v4f* __restrict__ src = (const v4f*)in;
  v4h* __restrict__ dst = (v4h*)out;
  for (; i < n4; i += stride) {
    v4f f = src[i];
    v4h h;
#pragma unroll
    for (int k = 0; k < 4; ++k) h[k] = (h16)f[k];
    dst[i] = h;
  }
}

// z_q gather (f32 codebook -> f16 decoder input) + VQ squared-error sum.
__global__ __launch_bounds__(256) void gather_vq_kernel(
    const float* __restrict__ pool, const float* __restrict__ ze32,
    const int* __restrict__ zidx, h16* __restrict__ zqh,
    float* __restrict__ acc, size_t n) {  // n = R*256
  size_t i = (size_t)blockIdx.x * blockDim.x + threadIdx.x;
  const size_t stride = (size_t)gridDim.x * blockDim.x;
  float s = 0.f;
  for (; i < n; i += stride) {
    const size_t bn = i >> 8;
    const int d = (int)(i & 255);
    const int k = zidx[bn];
    const float q = pool[(size_t)k * 256 + d];
    zqh[i] = (h16)q;
    const float diff = ze32[i] - q;
    s += diff * diff;
  }
  wave_atomic_add(acc + 1, s);
}

__global__ __launch_bounds__(256) void sq_diff_reduce_kernel(
    const float* __restrict__ a, const float* __restrict__ b,
    float* __restrict__ acc, size_t n) {
  size_t i = (size_t)blockIdx.x * blockDim.x + threadIdx.x;
  const size_t stride = (size_t)gridDim.x * blockDim.x;
  float s = 0.f;
  for (; i < n; i += stride) { float d = a[i] - b[i]; s += d * d; }
  wave_atomic_add(acc + 0, s);
}

__global__ __launch_bounds__(256) void onehot_kernel(
    const int* __restrict__ zidx, int* __restrict__ out, size_t n) {  // n = R*1024
  size_t i = (size_t)blockIdx.x * blockDim.x + threadIdx.x;
  const size_t stride = (size_t)gridDim.x * blockDim.x;
  for (; i < n; i += stride) {
    const size_t bn = i >> 10;
    const int c = (int)(i & 1023);
    out[i] = (zidx[bn] == c) ? 1 : 0;
  }
}

__global__ void finalize_loss_kernel(const float* __restrict__ acc,
                                     float* __restrict__ out) {
  const float meanA = acc[0] / (4096.0f * 3072.0f);        // recon MSE
  const float meanB = acc[1] / (4096.0f * 32.0f * 256.0f); // VQ MSE
  out[0] = (meanA + 1.25f * meanB) / 4096.0f;
}

// ---------------------------------------------------------------------------
// Host launcher
// ---------------------------------------------------------------------------
extern "C" void kernel_launch(void* const* d_in, const int* in_sizes, int n_in,
                              void* d_out, int out_size, void* d_ws, size_t ws_size,
                              hipStream_t stream) {
  (void)in_sizes; (void)n_in; (void)out_size; (void)ws_size;

  const float* x    = (const float*)d_in[0];   // [4096, 3072]
  const float* pool = (const float*)d_in[1];   // [1024, 256]
  const float* ew1  = (const float*)d_in[2];   const float* eb1 = (const float*)d_in[3];
  const float* ew2  = (const float*)d_in[4];   const float* eb2 = (const float*)d_in[5];
  const float* ew3  = (const float*)d_in[6];   const float* eb3 = (const float*)d_in[7];
  const float* dw1  = (const float*)d_in[8];   const float* db1 = (const float*)d_in[9];
  const float* dw2  = (const float*)d_in[10];  const float* db2 = (const float*)d_in[11];
  const float* dw3  = (const float*)d_in[12];  const float* db3 = (const float*)d_in[13];

  // Output layout: x_pred f32 [4096*3072] | z_discrete i32 [4096*32*1024] | loss f32
  float* xpred = (float*)d_out;
  int*   zdisc = (int*)(xpred + (size_t)4096 * 3072);
  float* lossp = (float*)d_out + ((size_t)4096 * 3072 + (size_t)4096 * 32 * 1024);

  // Workspace carve-out
  char* ws = (char*)d_ws;
  size_t off = 0;
  auto take = [&](size_t bytes) -> char* {
    char* p = ws + off;
    off = (off + bytes + 255) & ~(size_t)255;
    return p;
  };
  h16*   xh    = (h16*)take((size_t)4096 * 3072 * 2);
  h16*   ew1h  = (h16*)take((size_t)4096 * 3072 * 2);
  h16*   ew2h  = (h16*)take((size_t)4096 * 4096 * 2);
  h16*   ew3h  = (h16*)take((size_t)8192 * 4096 * 2);
  h16*   dw1h  = (h16*)take((size_t)4096 * 8192 * 2);
  h16*   dw2h  = (h16*)take((size_t)4096 * 4096 * 2);
  h16*   dw3h  = (h16*)take((size_t)3072 * 4096 * 2);
  h16*   poolh = (h16*)take((size_t)1024 * 256 * 2);
  h16*   actA  = (h16*)take((size_t)4096 * 4096 * 2);   // h1, later g1
  h16*   actB  = (h16*)take((size_t)4096 * 4096 * 2);   // h2, later g2
  float* ze32  = (float*)take((size_t)4096 * 8192 * 4); // z_e f32 (for loss)
  h16*   zeqh  = (h16*)take((size_t)4096 * 8192 * 2);   // z_e f16, reused as z_q f16
  int*   zidx  = (int*)take((size_t)131072 * 4);
  float* acc   = (float*)take(256);                     // [0]=recon, [1]=vq

  hipMemsetAsync(acc, 0, 2 * sizeof(float), stream);

  // f32 -> f16 staging (inputs, weights, codebook), vectorized x4
  cvt_f32_to_f16<<<2048, 256, 0, stream>>>(x,    xh,    (size_t)4096 * 3072 / 4);
  cvt_f32_to_f16<<<2048, 256, 0, stream>>>(ew1,  ew1h,  (size_t)4096 * 3072 / 4);
  cvt_f32_to_f16<<<2048, 256, 0, stream>>>(ew2,  ew2h,  (size_t)4096 * 4096 / 4);
  cvt_f32_to_f16<<<4096, 256, 0, stream>>>(ew3,  ew3h,  (size_t)8192 * 4096 / 4);
  cvt_f32_to_f16<<<4096, 256, 0, stream>>>(dw1,  dw1h,  (size_t)4096 * 8192 / 4);
  cvt_f32_to_f16<<<2048, 256, 0, stream>>>(dw2,  dw2h,  (size_t)4096 * 4096 / 4);
  cvt_f32_to_f16<<<2048, 256, 0, stream>>>(dw3,  dw3h,  (size_t)3072 * 4096 / 4);
  cvt_f32_to_f16<<<256,  256, 0, stream>>>(pool, poolh, (size_t)1024 * 256 / 4);

  const dim3 blk(256);
  // Encoder (block tile 128x128)
  gemm_wmma_kernel<1, false, true><<<dim3(4096 / 128, 4096 / 128), blk, 0, stream>>>(
      xh, ew1h, eb1, nullptr, actA, 4096, 4096, 3072);
  gemm_wmma_kernel<1, false, true><<<dim3(4096 / 128, 4096 / 128), blk, 0, stream>>>(
      actA, ew2h, eb2, nullptr, actB, 4096, 4096, 4096);
  gemm_wmma_kernel<0, true, true><<<dim3(8192 / 128, 4096 / 128), blk, 0, stream>>>(
      actB, ew3h, eb3, ze32, zeqh, 4096, 8192, 4096);

  // Codebook scoring + argmax (131072 rows, 128 per block)
  score_argmax_kernel<<<1024, 256, 0, stream>>>(zeqh, poolh, zidx, 131072);

  // z_q gather (forward value of straight-through z_ss) + VQ loss
  gather_vq_kernel<<<8192, 256, 0, stream>>>(pool, ze32, zidx, zeqh, acc,
                                             (size_t)131072 * 256);

  // Decoder
  gemm_wmma_kernel<1, false, true><<<dim3(4096 / 128, 4096 / 128), blk, 0, stream>>>(
      zeqh, dw1h, db1, nullptr, actA, 4096, 4096, 8192);
  gemm_wmma_kernel<1, false, true><<<dim3(4096 / 128, 4096 / 128), blk, 0, stream>>>(
      actA, dw2h, db2, nullptr, actB, 4096, 4096, 4096);
  gemm_wmma_kernel<2, true, false><<<dim3(3072 / 128, 4096 / 128), blk, 0, stream>>>(
      actB, dw3h, db3, xpred, nullptr, 4096, 3072, 4096);

  // Losses + one-hot output
  sq_diff_reduce_kernel<<<4096, 256, 0, stream>>>(x, xpred, acc, (size_t)4096 * 3072);
  onehot_kernel<<<8192, 256, 0, stream>>>(zidx, zdisc, (size_t)131072 * 1024);
  finalize_loss_kernel<<<1, 1, 0, stream>>>(acc, lossp);
}
